// My_SimpleRNN_Cell_74045236183055
// MI455X (gfx1250) — matrix operand order repaired and checked by reading, not compiled, not run on hardware
//
#include <hip/hip_runtime.h>
#include <hip/hip_bf16.h>

#define BB 128
#define TT 1024
#define DD 128
#define HH 128

typedef __attribute__((ext_vector_type(16))) __bf16 v16bf;
typedef __attribute__((ext_vector_type(8)))  float  v8f;

union Frag16 {
  v16bf v;
  __bf16 h[16];
  unsigned short us[16];
  uint4 q[2];
};

struct bf4x { __bf16 x, y, z, w; };

// Let the backend pick the native f32->bf16 convert (RNE); on gfx1250 this
// should select v_cvt_*bf16_f32 instead of a 3-op software RNE sequence.
__device__ __forceinline__ __bf16 cvt_bf16(float f) { return (__bf16)f; }

// Padded bf16 tile stride: 128 + 8 elems -> 272B rows; b128 fragment reads
// across lanes hit banks (4*m + k/2) mod 64 -> conflict-free.
#define SH_STRIDE 136

// ---------------------------------------------------------------------------
// Kernel 1: xu = x @ U + bu   ([B*T,128] @ [128,128] -> written into d_out)
// Persistent workgroups. Staging is software-pipelined: next tile's two
// global_load_b128 per thread are issued while the current tile's WMMAs and
// stores run; fp32->bf16 conversion happens once, into padded LDS.
// ---------------------------------------------------------------------------
__global__ __launch_bounds__(256) void xu_gemm_kernel(
    const float* __restrict__ x, const float* __restrict__ U,
    const float* __restrict__ bu, float* __restrict__ xu)
{
  __shared__ __align__(16) __bf16 sX[16 * SH_STRIDE];

  const int lane = threadIdx.x & 31;
  const int wave = threadIdx.x >> 5;          // N tile 0..7
  const int half = lane >> 4;                 // 0: lanes 0-15, 1: lanes 16-31
  const int m    = lane & 15;
  const int nCol = (wave << 4) + m;

  // B fragments of U (bf16), loaded once per wave.
  // 16-bit B layout: lane = N col; lanes 0-15 hold K=kc*32+0..7,16..23, lanes 16-31 K=+8..15,24..31.
  Frag16 uf[4];
#pragma unroll
  for (int kc = 0; kc < 4; ++kc) {
#pragma unroll
    for (int e = 0; e < 16; ++e) {
      int k = kc * 32 + half * 8 + (e < 8 ? e : e + 8);
      uf[kc].h[e] = cvt_bf16(U[k * HH + nCol]);
    }
  }
  const float bias = bu[nCol];

  const int numTiles      = (BB * TT) / 16;                 // 8192
  const int tilesPerBlock = numTiles / gridDim.x;

  // Each thread stages exactly 2 of the 512 float4 chunks of a 16x128 tile.
  const int c0 = threadIdx.x;        // chunk 0..255
  const int c1 = threadIdx.x + 256;  // chunk 256..511
  const int r0 = c0 >> 5, q0 = (c0 & 31) * 4;
  const int r1 = c1 >> 5, q1 = (c1 & 31) * 4;

  int rowBase = blockIdx.x * tilesPerBlock * 16;
  float4 fa = *(const float4*)&x[(rowBase + r0) * DD + q0];
  float4 fb = *(const float4*)&x[(rowBase + r1) * DD + q1];

  for (int it = 0; it < tilesPerBlock; ++it) {
    // Stage current tile (convert fp32 -> bf16 once, 8B stores).
    {
      bf4x pa = { cvt_bf16(fa.x), cvt_bf16(fa.y), cvt_bf16(fa.z), cvt_bf16(fa.w) };
      bf4x pb = { cvt_bf16(fb.x), cvt_bf16(fb.y), cvt_bf16(fb.z), cvt_bf16(fb.w) };
      *(bf4x*)&sX[r0 * SH_STRIDE + q0] = pa;
      *(bf4x*)&sX[r1 * SH_STRIDE + q1] = pb;
    }
    __syncthreads();

    // Issue next tile's loads now; they complete behind the WMMAs/stores.
    if (it + 1 < tilesPerBlock) {
      const int nb = rowBase + 16;
      fa = *(const float4*)&x[(nb + r0) * DD + q0];
      fb = *(const float4*)&x[(nb + r1) * DD + q1];
      __builtin_prefetch(&x[(nb + 16) * DD + threadIdx.x * 8], 0, 1);
    }

    // A fragments: lane(0-15) M=m holds K = kb+0..7 (elems 0-7) and
    // kb+16..23 (elems 8-15); lanes 16-31 use kb = kc*32 + 8.
    v8f acc0 = {}, acc1 = {};
#pragma unroll
    for (int kc = 0; kc < 4; ++kc) {
      Frag16 af;
      const int kb = kc * 32 + half * 8;
      af.q[0] = *(const uint4*)&sX[m * SH_STRIDE + kb];
      af.q[1] = *(const uint4*)&sX[m * SH_STRIDE + kb + 16];
      if (kc & 1)
        acc1 = __builtin_amdgcn_wmma_f32_16x16x32_bf16(false, af.v, false, uf[kc].v,
                                                       (short)0, acc1, false, false);
      else
        acc0 = __builtin_amdgcn_wmma_f32_16x16x32_bf16(false, af.v, false, uf[kc].v,
                                                       (short)0, acc0, false, false);
    }
    __syncthreads();

    // D layout: elem r -> row rowBase + r + half*8, col nCol.
#pragma unroll
    for (int r = 0; r < 8; ++r) {
      xu[(rowBase + r + half * 8) * HH + nCol] = acc0[r] + acc1[r] + bias;
    }
    rowBase += 16;
  }
}

// ---------------------------------------------------------------------------
// Kernel 2: sequential scan.
//   st = tanh(xu_t + h @ W + bw);  o = tanh(st @ V + bv);  h <- st
// 8 WGs (one per 16-row batch tile) x 8 waves (one per 16-col N tile).
// Hidden state in LDS (bf16, padded rows). xu read from / o written to d_out.
//
// Fragments of st(t) are loaded ONCE per step and feed BOTH the output GEMM
// st(t)@V and the next-step recurrence GEMM st(t)@W: 8 WMMAs in 4 independent
// 2-deep chains, 2 barriers/step, halved DS traffic.
// ---------------------------------------------------------------------------
__global__ __launch_bounds__(256) void rnn_scan_kernel(
    const float* __restrict__ h0, const float* __restrict__ W,
    const float* __restrict__ V,  const float* __restrict__ bw,
    const float* __restrict__ bv, float* __restrict__ out)
{
  __shared__ __align__(16) __bf16 sH[16 * SH_STRIDE];

  const int lane = threadIdx.x & 31;
  const int wave = threadIdx.x >> 5;
  const int half = lane >> 4;
  const int m    = lane & 15;
  const int nCol = (wave << 4) + m;
  const int rb   = blockIdx.x * 16;  // batch-row base

  // Weight B fragments (bf16), once per wave.
  Frag16 wf[4], vf[4];
#pragma unroll
  for (int kc = 0; kc < 4; ++kc) {
#pragma unroll
    for (int e = 0; e < 16; ++e) {
      int k = kc * 32 + half * 8 + (e < 8 ? e : e + 8);
      wf[kc].h[e] = cvt_bf16(W[k * HH + nCol]);
      vf[kc].h[e] = cvt_bf16(V[k * HH + nCol]);
    }
  }
  const float bwn = bw[nCol];
  const float bvn = bv[nCol];

  // Init hidden state LDS from h0 (bf16).
  for (int i = threadIdx.x; i < 16 * HH; i += 256) {
    int row = i >> 7, col = i & (HH - 1);
    sH[row * SH_STRIDE + col] = cvt_bf16(h0[(rb + row) * HH + col]);
  }
  __syncthreads();

  // ---- Prologue: fragments of h(0) and the t=0 recurrence GEMM h0@W. ----
  Frag16 f[4];
#pragma unroll
  for (int kc = 0; kc < 4; ++kc) {
    const int kb = kc * 32 + half * 8;
    f[kc].q[0] = *(const uint4*)&sH[m * SH_STRIDE + kb];
    f[kc].q[1] = *(const uint4*)&sH[m * SH_STRIDE + kb + 16];
  }
  v8f w0 = {}, w1 = {};
  w0 = __builtin_amdgcn_wmma_f32_16x16x32_bf16(false, f[0].v, false, wf[0].v, (short)0, w0, false, false);
  w1 = __builtin_amdgcn_wmma_f32_16x16x32_bf16(false, f[1].v, false, wf[1].v, (short)0, w1, false, false);
  w0 = __builtin_amdgcn_wmma_f32_16x16x32_bf16(false, f[2].v, false, wf[2].v, (short)0, w0, false, false);
  w1 = __builtin_amdgcn_wmma_f32_16x16x32_bf16(false, f[3].v, false, wf[3].v, (short)0, w1, false, false);

  // xu tile for t=0 (rows rb + r + half*8, col nCol).
  float xu_cur[8];
#pragma unroll
  for (int r = 0; r < 8; ++r)
    xu_cur[r] = out[((rb + r + half * 8) * TT + 0) * HH + nCol];

  for (int t = 0; t < TT; ++t) {
    // st(t) = tanh(h(t)@W + xu_t + bw); h(t)@W was computed last iteration.
    float st[8];
#pragma unroll
    for (int r = 0; r < 8; ++r)
      st[r] = tanhf(w0[r] + w1[r] + xu_cur[r] + bwn);

    __syncthreads();  // all waves done reading fragments of h(t) from LDS

    // Publish st(t) as the new hidden state (bf16).
#pragma unroll
    for (int r = 0; r < 8; ++r)
      sH[(r + half * 8) * SH_STRIDE + nCol] = cvt_bf16(st[r]);
    __syncthreads();  // st(t) visible to all waves

    // Load fragments of st(t) ONCE; they feed both st@V and st@W.
#pragma unroll
    for (int kc = 0; kc < 4; ++kc) {
      const int kb = kc * 32 + half * 8;
      f[kc].q[0] = *(const uint4*)&sH[m * SH_STRIDE + kb];
      f[kc].q[1] = *(const uint4*)&sH[m * SH_STRIDE + kb + 16];
    }

    // Prefetch next xu tile while the WMMAs run.
    float xu_nxt[8];
    {
      const int tn = (t + 1 < TT) ? (t + 1) : t;
#pragma unroll
      for (int r = 0; r < 8; ++r)
        xu_nxt[r] = out[((rb + r + half * 8) * TT + tn) * HH + nCol];
    }

    // 8 WMMAs in 4 independent 2-deep accumulation chains.
    v8f o0 = {}, o1 = {};
    w0 = v8f{};
    w1 = v8f{};
    o0 = __builtin_amdgcn_wmma_f32_16x16x32_bf16(false, f[0].v, false, vf[0].v, (short)0, o0, false, false);
    w0 = __builtin_amdgcn_wmma_f32_16x16x32_bf16(false, f[0].v, false, wf[0].v, (short)0, w0, false, false);
    o1 = __builtin_amdgcn_wmma_f32_16x16x32_bf16(false, f[1].v, false, vf[1].v, (short)0, o1, false, false);
    w1 = __builtin_amdgcn_wmma_f32_16x16x32_bf16(false, f[1].v, false, wf[1].v, (short)0, w1, false, false);
    o0 = __builtin_amdgcn_wmma_f32_16x16x32_bf16(false, f[2].v, false, vf[2].v, (short)0, o0, false, false);
    w0 = __builtin_amdgcn_wmma_f32_16x16x32_bf16(false, f[2].v, false, wf[2].v, (short)0, w0, false, false);
    o1 = __builtin_amdgcn_wmma_f32_16x16x32_bf16(false, f[3].v, false, vf[3].v, (short)0, o1, false, false);
    w1 = __builtin_amdgcn_wmma_f32_16x16x32_bf16(false, f[3].v, false, wf[3].v, (short)0, w1, false, false);

    // o(t) = tanh(st@V + bv); overwrite the consumed xu slice in-place.
#pragma unroll
    for (int r = 0; r < 8; ++r) {
      float o = tanhf(o0[r] + o1[r] + bvn);
      out[((rb + r + half * 8) * TT + t) * HH + nCol] = o;
    }

#pragma unroll
    for (int r = 0; r < 8; ++r) xu_cur[r] = xu_nxt[r];
  }
}

// ---------------------------------------------------------------------------
extern "C" void kernel_launch(void* const* d_in, const int* in_sizes, int n_in,
                              void* d_out, int out_size, void* d_ws, size_t ws_size,
                              hipStream_t stream) {
  const float* x  = (const float*)d_in[0];
  const float* h0 = (const float*)d_in[1];
  const float* U  = (const float*)d_in[2];
  const float* W  = (const float*)d_in[3];
  const float* V  = (const float*)d_in[4];
  const float* bu = (const float*)d_in[5];
  const float* bw = (const float*)d_in[6];
  const float* bv = (const float*)d_in[7];
  float* out = (float*)d_out;

  // Phase 1: xu = x@U + bu written into d_out (same [B,T,H] shape).
  xu_gemm_kernel<<<1024, 256, 0, stream>>>(x, U, bu, out);
  // Phase 2: sequential scan; reads xu from d_out, overwrites with o in-place.
  rnn_scan_kernel<<<BB / 16, 256, 0, stream>>>(h0, W, V, bw, bv, out);
}
